// Attention_45406394253508
// MI455X (gfx1250) — matrix operand order repaired and checked by reading, not compiled, not run on hardware
//
#include <hip/hip_runtime.h>
#include <hip/hip_bf16.h>

typedef __bf16 bf16_t;
typedef __attribute__((ext_vector_type(16))) __bf16 v16bf;
typedef __attribute__((ext_vector_type(8)))  __bf16 v8bf;
typedef __attribute__((ext_vector_type(8)))  float  v8f;

static constexpr int N_SEQ   = 2048;
static constexpr int DIM     = 1024;
static constexpr int HEADS   = 16;
static constexpr int DHEAD   = 64;
static constexpr int INNER   = 1024;          // HEADS*DHEAD
static constexpr size_t NN2  = (size_t)N_SEQ * N_SEQ;   // 4M per head
static constexpr float ALPHA = 0.45f;
static constexpr float SCALE = 0.125f;        // 64^-0.5

enum { MODE_PROJ = 0, MODE_PROJ_T = 1, MODE_SCORES = 2, MODE_AV = 3, MODE_OUT = 4 };

// ---------------------------------------------------------------------------
// WMMA helper (D = A(16x32 bf16) * B(32x16 bf16) + C(f32))
// ---------------------------------------------------------------------------
__device__ inline v8f wmma_bf16(v16bf a, v16bf b, v8f c) {
  return __builtin_amdgcn_wmma_f32_16x16x32_bf16(
      /*neg_a=*/false, a, /*neg_b=*/false, b,
      /*c_mod=*/(short)0, c, /*reuse_a=*/false, /*reuse_b=*/false);
}

static constexpr int BM = 64, BN = 64, BK = 32;

// A fragment: ISA 16-bit A 16x32 layout.
// lane l: row = m0 + (l&15); lanes 0-15 hold K {0..7,16..23}, lanes 16-31 hold
// K {8..15,24..31}  ->  two contiguous 8-element (16B) ds reads per lane.
__device__ inline v16bf load_a_frag(const bf16_t* As, int m0, int lane) {
  int row = m0 + (lane & 15);
  int kh  = (lane >> 4) * 8;
  const bf16_t* p = As + row * BK;
  v8bf lo = *reinterpret_cast<const v8bf*>(p + kh);
  v8bf hi = *reinterpret_cast<const v8bf*>(p + 16 + kh);
  return __builtin_shufflevector(lo, hi, 0,1,2,3,4,5,6,7,8,9,10,11,12,13,14,15);
}

// B fragment: 32x16 bf16; lane l: col n = n0+(l&15); lanes 0-15 hold K {0..7,16..23},
// lanes 16-31 hold K {8..15,24..31} -> two contiguous 16B ds reads per lane.
// Bs is stored n-major: Bs[n][k].
__device__ inline v16bf load_b_frag(const bf16_t* Bs, int n0, int lane) {
  int n  = n0 + (lane & 15);
  int kh = (lane >> 4) * 8;
  const bf16_t* p = Bs + n * BK;
  v8bf lo = *reinterpret_cast<const v8bf*>(p + kh);
  v8bf hi = *reinterpret_cast<const v8bf*>(p + 16 + kh);
  return __builtin_shufflevector(lo, hi, 0,1,2,3,4,5,6,7,8,9,10,11,12,13,14,15);
}

// ---------------------------------------------------------------------------
// Templated WMMA GEMM. All B operands are n-major in global memory
// (weights pre-transposed; v emitted transposed by MODE_PROJ_T), so staging
// is one uniform fully-vectorized path: b128 global loads -> b128 LDS stores.
//   MODE_PROJ   : Cb[M x N](bf16, row-major) = A @ B                   (z=1)
//   MODE_PROJ_T : Cb[N x M](bf16, C transposed) = A @ B               (z=1)
//   MODE_SCORES : blended(f32) = a*s*(q_h @ k_h^T) + (1-a)*h          (z=16)
//   MODE_AV     : outheads(bf16) = attn_h @ v_h                       (z=16)
//   MODE_OUT    : Cf[M x N](f32) = A @ B + bias                       (z=1)
// ---------------------------------------------------------------------------
template <int MODE, int M, int N, int K>
__global__ __launch_bounds__(128)
void gemm_wmma(const bf16_t* __restrict__ A, const bf16_t* __restrict__ B,
               bf16_t* __restrict__ Cb, float* __restrict__ Cf,
               const float* __restrict__ Haux, const float* __restrict__ bias) {
  __shared__ alignas(64) bf16_t As[BM * BK];
  __shared__ alignas(64) bf16_t Bs[BN * BK];

  const int tid  = threadIdx.x;
  const int lane = tid & 31;
  const int wave = tid >> 5;
  const int wr   = wave >> 1;          // 0..1 : 32-row strip inside 64-row tile
  const int wc   = wave & 1;           // 0..1 : 32-col strip inside 64-col tile
  const int rowBase = blockIdx.y * BM;
  const int colBase = blockIdx.x * BN;
  const int head    = blockIdx.z;

  // element index of A(r, k); all layouts k-fastest, 8-element aligned
  auto a_index = [&](int r, int k) -> size_t {
    if constexpr (MODE == MODE_SCORES)
      return (size_t)r * INNER + head * DHEAD + k;                 // q flat
    else if constexpr (MODE == MODE_AV)
      return (size_t)head * NN2 + (size_t)r * N_SEQ + k;           // attn[h]
    else
      return (size_t)r * (size_t)K + k;                            // row-major
  };
  // element index of B as Bs[n][k]; all layouts n-major, k-fastest
  auto b_index = [&](int n, int k) -> size_t {
    if constexpr (MODE == MODE_SCORES)
      return (size_t)(colBase + n) * INNER + head * DHEAD + k;     // k rows
    else if constexpr (MODE == MODE_AV)
      return (size_t)(head * DHEAD + colBase + n) * N_SEQ + k;     // v^T
    else
      return (size_t)(colBase + n) * (size_t)K + k;                // W^T
  };

  v8f acc[2][2] = {};

  for (int k0 = 0; k0 < K; k0 += BK) {
    // ---- vectorized staging: 256 chunks of 8 bf16 (16B), 2 per thread ----
#pragma unroll
    for (int p = 0; p < 2; p++) {
      int c  = tid + p * 128;          // 0..255
      int r  = c >> 2;                 // 0..63
      int ko = (c & 3) * 8;            // 0,8,16,24
      *reinterpret_cast<v8bf*>(As + r * BK + ko) =
          *reinterpret_cast<const v8bf*>(A + a_index(rowBase + r, k0 + ko));
      *reinterpret_cast<v8bf*>(Bs + r * BK + ko) =
          *reinterpret_cast<const v8bf*>(B + b_index(r, k0 + ko));
    }
    if (k0 + BK < K) {                 // next K tile -> global_prefetch_b8
      __builtin_prefetch(A + a_index(rowBase + (tid >> 1), k0 + BK), 0, 1);
      __builtin_prefetch(B + b_index(tid >> 1, k0 + BK), 0, 1);
    }
    __syncthreads();

    v16bf a0 = load_a_frag(As, wr * 32,      lane);
    v16bf a1 = load_a_frag(As, wr * 32 + 16, lane);
    v16bf b0 = load_b_frag(Bs, wc * 32,      lane);
    v16bf b1 = load_b_frag(Bs, wc * 32 + 16, lane);
    acc[0][0] = wmma_bf16(a0, b0, acc[0][0]);
    acc[0][1] = wmma_bf16(a0, b1, acc[0][1]);
    acc[1][0] = wmma_bf16(a1, b0, acc[1][0]);
    acc[1][1] = wmma_bf16(a1, b1, acc[1][1]);
    __syncthreads();
  }

  // ---- epilogue: f32 16x16 C layout: lane -> N=lane&15, M = v + (lane>>4)*8
#pragma unroll
  for (int im = 0; im < 2; im++)
#pragma unroll
    for (int in_ = 0; in_ < 2; in_++)
#pragma unroll
      for (int v = 0; v < 8; v++) {
        int row = rowBase + wr * 32 + im * 16 + ((lane >> 4) << 3) + v;
        int col = colBase + wc * 32 + in_ * 16 + (lane & 15);
        float c = acc[im][in_][v];
        if constexpr (MODE == MODE_SCORES) {
          size_t idx = (size_t)head * NN2 + (size_t)row * N_SEQ + col;
          Cf[idx] = (ALPHA * SCALE) * c + (1.0f - ALPHA) * Haux[idx];
        } else if constexpr (MODE == MODE_AV) {
          Cb[(size_t)row * INNER + head * DHEAD + col] = (bf16_t)c;
        } else if constexpr (MODE == MODE_OUT) {
          Cf[(size_t)row * (size_t)N + col] = c + bias[col];
        } else if constexpr (MODE == MODE_PROJ_T) {
          Cb[(size_t)col * (size_t)M + row] = (bf16_t)c;   // C transposed
        } else {
          Cb[(size_t)row * (size_t)N + col] = (bf16_t)c;
        }
      }
}

// ---------------------------------------------------------------------------
// LayerNorm (f32 in, bf16 out), one block per row
// ---------------------------------------------------------------------------
__global__ __launch_bounds__(256)
void layernorm_bf16(const float* __restrict__ x, const float* __restrict__ g,
                    const float* __restrict__ b, bf16_t* __restrict__ xn) {
  __shared__ float rs[256], rs2[256];
  const int row = blockIdx.x;
  const float* xr = x + (size_t)row * DIM;
  float s = 0.f, s2 = 0.f;
  for (int i = threadIdx.x; i < DIM; i += 256) {
    float v = xr[i]; s += v; s2 += v * v;
  }
  rs[threadIdx.x] = s; rs2[threadIdx.x] = s2;
  __syncthreads();
  for (int st = 128; st > 0; st >>= 1) {
    if (threadIdx.x < st) {
      rs[threadIdx.x]  += rs[threadIdx.x + st];
      rs2[threadIdx.x] += rs2[threadIdx.x + st];
    }
    __syncthreads();
  }
  const float mean = rs[0] * (1.0f / DIM);
  const float var  = rs2[0] * (1.0f / DIM) - mean * mean;
  const float rstd = rsqrtf(var + 1e-5f);
  for (int i = threadIdx.x; i < DIM; i += 256) {
    float v = (xr[i] - mean) * rstd * g[i] + b[i];
    xn[(size_t)row * DIM + i] = (bf16_t)v;
  }
}

// ---------------------------------------------------------------------------
// f32 -> bf16 transposed weight conversion: d[n*K + k] = s[k*lds + coloff + n]
// One-time ~24 MB total (16 MB f32 read, 8 MB bf16 written); coalescing secondary.
// ---------------------------------------------------------------------------
__global__ void f32_to_bf16_T(const float* __restrict__ s, bf16_t* __restrict__ d,
                              int K, int N, int coloff, int lds) {
  for (int idx = blockIdx.x * blockDim.x + threadIdx.x; idx < K * N;
       idx += gridDim.x * blockDim.x) {
    int n = idx / K, k = idx - n * K;
    d[idx] = (bf16_t)s[(size_t)k * lds + coloff + n];
  }
}

// ---------------------------------------------------------------------------
// Talking-heads: dots = mix_pre^T over heads, softmax over j, mix_post,
// emit bf16 attention probs. One block per row i; dots recomputed per pass
// (per-row working set = 16*2048*4B = 128 KB, re-reads stay in L2).
// ---------------------------------------------------------------------------
__global__ __launch_bounds__(256)
void softmax_mix(const float* __restrict__ blended, const float* __restrict__ mix_pre,
                 const float* __restrict__ mix_post, bf16_t* __restrict__ attn) {
  __shared__ float s_mp[256], s_mq[256];
  __shared__ float red[16 * 260];
  __shared__ float s_max[16], s_inv[16];

  const int t = threadIdx.x;
  const int i = blockIdx.x;
  const size_t rowoff = (size_t)i * N_SEQ;

  s_mp[t] = mix_pre[t];
  s_mq[t] = mix_post[t];
  __syncthreads();

  auto dots_at = [&](int j, float* d) {
    float bl[16];
#pragma unroll
    for (int h = 0; h < 16; h++) bl[h] = blended[(size_t)h * NN2 + rowoff + j];
#pragma unroll
    for (int g = 0; g < 16; g++) {
      float a = 0.f;
#pragma unroll
      for (int h = 0; h < 16; h++) a = fmaf(bl[h], s_mp[h * 16 + g], a);
      d[g] = a;
    }
  };

  // pass 1: per-head max
  float gmax[16];
#pragma unroll
  for (int g = 0; g < 16; g++) gmax[g] = -3.4e38f;
  for (int jj = 0; jj < N_SEQ / 256; jj++) {
    float d[16];
    dots_at(t + jj * 256, d);
#pragma unroll
    for (int g = 0; g < 16; g++) gmax[g] = fmaxf(gmax[g], d[g]);
  }
#pragma unroll
  for (int g = 0; g < 16; g++) red[g * 260 + t] = gmax[g];
  __syncthreads();
  if (t < 16) {
    float m = -3.4e38f;
    for (int k = 0; k < 256; k++) m = fmaxf(m, red[t * 260 + k]);
    s_max[t] = m;
  }
  __syncthreads();

  // pass 2: per-head sum of exp
  float gsum[16];
#pragma unroll
  for (int g = 0; g < 16; g++) gsum[g] = 0.f;
  for (int jj = 0; jj < N_SEQ / 256; jj++) {
    float d[16];
    dots_at(t + jj * 256, d);
#pragma unroll
    for (int g = 0; g < 16; g++) gsum[g] += __expf(d[g] - s_max[g]);
  }
#pragma unroll
  for (int g = 0; g < 16; g++) red[g * 260 + t] = gsum[g];
  __syncthreads();
  if (t < 16) {
    float s = 0.f;
    for (int k = 0; k < 256; k++) s += red[t * 260 + k];
    s_inv[t] = 1.0f / s;
  }
  __syncthreads();

  // pass 3: probs + post-mix -> bf16 attn[g][i][j]
  for (int jj = 0; jj < N_SEQ / 256; jj++) {
    const int j = t + jj * 256;
    float d[16], p[16];
    dots_at(j, d);
#pragma unroll
    for (int h = 0; h < 16; h++) p[h] = __expf(d[h] - s_max[h]) * s_inv[h];
#pragma unroll
    for (int g = 0; g < 16; g++) {
      float a = 0.f;
#pragma unroll
      for (int h = 0; h < 16; h++) a = fmaf(p[h], s_mq[h * 16 + g], a);
      attn[(size_t)g * NN2 + rowoff + j] = (bf16_t)a;
    }
  }
}

// ---------------------------------------------------------------------------
// host launcher
// ---------------------------------------------------------------------------
extern "C" void kernel_launch(void* const* d_in, const int* in_sizes, int n_in,
                              void* d_out, int out_size, void* d_ws, size_t ws_size,
                              hipStream_t stream) {
  const float* x       = (const float*)d_in[0];
  const float* h       = (const float*)d_in[1];
  const float* ln_g    = (const float*)d_in[2];
  const float* ln_b    = (const float*)d_in[3];
  const float* Wq      = (const float*)d_in[4];
  const float* Wkv     = (const float*)d_in[5];
  const float* mix_pre = (const float*)d_in[6];
  const float* mix_post= (const float*)d_in[7];
  const float* Wout    = (const float*)d_in[8];
  const float* bout    = (const float*)d_in[9];

  float* outp    = (float*)d_out;                       // (2048,1024)
  float* blended = outp + (size_t)N_SEQ * DIM;          // (16,2048,2048)

  // workspace carve-out (~156 MiB)
  char* w = (char*)d_ws;
  bf16_t* xn    = (bf16_t*)w; w += (size_t)N_SEQ * DIM * 2;
  bf16_t* WqT   = (bf16_t*)w; w += (size_t)DIM * DIM * 2;     // [n][k]
  bf16_t* WkT   = (bf16_t*)w; w += (size_t)DIM * DIM * 2;
  bf16_t* WvT   = (bf16_t*)w; w += (size_t)DIM * DIM * 2;
  bf16_t* WoT   = (bf16_t*)w; w += (size_t)DIM * DIM * 2;
  bf16_t* qb    = (bf16_t*)w; w += (size_t)N_SEQ * INNER * 2; // [j][h*64+d]
  bf16_t* kb    = (bf16_t*)w; w += (size_t)N_SEQ * INNER * 2; // [j][h*64+d]
  bf16_t* vt    = (bf16_t*)w; w += (size_t)INNER * N_SEQ * 2; // [h*64+d][j]
  bf16_t* ohb   = (bf16_t*)w; w += (size_t)N_SEQ * INNER * 2; // [j][h*64+d]
  bf16_t* attnb = (bf16_t*)w;                                 // [h][i][j] bf16

  dim3 blk(128, 1, 1);

  layernorm_bf16<<<N_SEQ, 256, 0, stream>>>(x, ln_g, ln_b, xn);
  f32_to_bf16_T<<<1024, 256, 0, stream>>>(Wq,  WqT, DIM, DIM, 0,    DIM);
  f32_to_bf16_T<<<1024, 256, 0, stream>>>(Wkv, WkT, DIM, DIM, 0,    2 * DIM);
  f32_to_bf16_T<<<1024, 256, 0, stream>>>(Wkv, WvT, DIM, DIM, DIM,  2 * DIM);
  f32_to_bf16_T<<<1024, 256, 0, stream>>>(Wout, WoT, DIM, DIM, 0,   DIM);

  // q = xn @ Wq ; k = xn @ Wk ; v^T = (xn @ Wv)^T
  gemm_wmma<MODE_PROJ, N_SEQ, DIM, DIM>
      <<<dim3(DIM / BN, N_SEQ / BM, 1), blk, 0, stream>>>(
          xn, WqT, qb, nullptr, nullptr, nullptr);
  gemm_wmma<MODE_PROJ, N_SEQ, DIM, DIM>
      <<<dim3(DIM / BN, N_SEQ / BM, 1), blk, 0, stream>>>(
          xn, WkT, kb, nullptr, nullptr, nullptr);
  gemm_wmma<MODE_PROJ_T, N_SEQ, DIM, DIM>
      <<<dim3(DIM / BN, N_SEQ / BM, 1), blk, 0, stream>>>(
          xn, WvT, vt, nullptr, nullptr, nullptr);
  // blended = ALPHA*SCALE*(q_h @ k_h^T) + (1-ALPHA)*h   (output #2)
  gemm_wmma<MODE_SCORES, N_SEQ, N_SEQ, DHEAD>
      <<<dim3(N_SEQ / BN, N_SEQ / BM, HEADS), blk, 0, stream>>>(
          qb, kb, nullptr, blended, h, nullptr);
  // talking-heads pre -> softmax -> talking-heads post
  softmax_mix<<<N_SEQ, 256, 0, stream>>>(blended, mix_pre, mix_post, attnb);
  // out_heads = attn_h @ v_h
  gemm_wmma<MODE_AV, N_SEQ, DHEAD, N_SEQ>
      <<<dim3(1, N_SEQ / BM, HEADS), blk, 0, stream>>>(
          attnb, vt, ohb, nullptr, nullptr, nullptr);
  // out = out_heads @ Wout + bout   (output #1)
  gemm_wmma<MODE_OUT, N_SEQ, DIM, INNER>
      <<<dim3(DIM / BN, N_SEQ / BM, 1), blk, 0, stream>>>(
          ohb, WoT, nullptr, outp, nullptr, bout);

  (void)in_sizes; (void)n_in; (void)out_size; (void)ws_size;
}
